// Decoder_46334107189745
// MI455X (gfx1250) — compile-verified, matrix operand order checked by reading, not run on hardware
//
#include <hip/hip_runtime.h>

// ---------------------------------------------------------------------------
// LSTM decoder (2 layers, B=64, H=1024, OUT=512, SEQ=256) with bf16 WMMA and
// async global->LDS staging (gfx1250 GLOBAL_LOAD_ASYNC_TO_LDS_B128 path).
// ---------------------------------------------------------------------------

typedef __bf16 bf16_t;
typedef __attribute__((ext_vector_type(16))) __bf16 v16bf;
typedef __attribute__((ext_vector_type(8)))  float  v8f;

#define BATCH 64
#define HID   1024
#define OUTD  512
#define SEQL  256
#define NGATE 4096   // 4*HID

union FragU { v16bf v; float4 q[2]; };

// Load a 16-element bf16 fragment for lane (m = lane&15, g = lane>>4):
// elements 0..7  <- p[0..7]   (K = g*8 .. g*8+7)
// elements 8..15 <- p[16..23] (K = 16+g*8 ..)
__device__ __forceinline__ v16bf frag_from(const bf16_t* p) {
    FragU u;
    u.q[0] = *reinterpret_cast<const float4*>(p);
    u.q[1] = *reinterpret_cast<const float4*>(p + 16);
    return u.v;
}

__device__ __forceinline__ float sigmoidf_(float x) {
    return 1.0f / (1.0f + __expf(-x));
}

// Async copy of 16 bytes/lane from global to LDS, tracked by ASYNCcnt.
// dst_lds: LDS byte address (addr[31:0] of the flat shared pointer).
__device__ __forceinline__ void async_ld_b128(uint32_t dst_lds, const void* src) {
    asm volatile("global_load_async_to_lds_b128 %0, %1, off"
                 :: "v"(dst_lds), "v"(src) : "memory");
}
__device__ __forceinline__ void wait_async_le1() {
    asm volatile("s_wait_asynccnt 0x1" ::: "memory");
}
__device__ __forceinline__ void wait_async_0() {
    asm volatile("s_wait_asynccnt 0x0" ::: "memory");
}

// ---------------------------------------------------------------------------
// Fused gate GEMM + LSTM cell.
// gates = x @ Wih^T + h @ Whh^T + (b_ih + b_hh), PyTorch gate order i,f,g,o.
// Grid: 32 blocks; block nb owns hidden columns [nb*32, nb*32+32) of all 4
// gates. 8 waves: wave -> (gate = wave/2, 16-col substrip); each computes
// 4 (M) x 1 (N) WMMA tiles. A tiles double-buffered in LDS via async loads;
// last iteration of each phase is peeled so the steady-state loop is
// branch-free and the cross-phase prefetch keeps the pipeline full.
// ---------------------------------------------------------------------------
__global__ __launch_bounds__(256)
void gate_cell_kernel(const bf16_t* __restrict__ x, int Kin,
                      const bf16_t* __restrict__ hin,
                      const bf16_t* __restrict__ Wih,   // (4096, Kin) row-major
                      const bf16_t* __restrict__ Whh,   // (4096, 1024)
                      const float*  __restrict__ b_ih,
                      const float*  __restrict__ b_hh,
                      float*        __restrict__ cst,   // (64, 1024) in/out
                      bf16_t*       __restrict__ hout)  // (64, 1024)
{
    // 32 KB: during GEMM, first 8 KB is a double-buffered 64x32 bf16 A tile;
    // afterwards the whole array holds activated gates [4][64][32] fp32.
    __shared__ __align__(16) float smem_f[4 * BATCH * 32];
    bf16_t* sA = reinterpret_cast<bf16_t*>(smem_f);

    const int tid  = threadIdx.x;
    const int wave = tid >> 5;
    const int lane = tid & 31;
    const int m16  = lane & 15;
    const int gsel = lane >> 4;
    const int nb   = blockIdx.x;          // 0..31
    const int gate = wave >> 1;           // 0..3
    const int nsub = (wave & 1) * 16;
    const int j0   = gate * HID + nb * 32 + nsub;  // first W row of B tile

    const uint32_t sA_lds = (uint32_t)(size_t)(void*)sA;   // LDS byte address

    v8f acc[4];
#pragma unroll
    for (int mt = 0; mt < 4; ++mt)
#pragma unroll
        for (int r = 0; r < 8; ++r) acc[mt][r] = 0.0f;

    const int arow = tid >> 2;   // 0..63
    const int acol = tid & 3;    // 16-byte chunk within 64-byte tile row

    // Per-thread base pointers (k advances by +32 elements along a row).
    const bf16_t* Ax  = x   + arow * Kin + acol * 8;
    const bf16_t* Ah  = hin + arow * HID + acol * 8;
    const bf16_t* Bw0 = Wih + (size_t)(j0 + m16) * Kin + gsel * 8;
    const bf16_t* Bw1 = Whh + (size_t)(j0 + m16) * HID + gsel * 8;
    const uint32_t ldst = sA_lds + tid * 16;

    auto compute_blk = [&](int cur, v16bf bf) {
        const bf16_t* abase = sA + cur * 2048 + m16 * 32 + gsel * 8;
#pragma unroll
        for (int mt = 0; mt < 4; ++mt) {
            v16bf af = frag_from(abase + mt * 16 * 32);
            acc[mt] = __builtin_amdgcn_wmma_f32_16x16x32_bf16(
                false, af, false, bf, (short)0, acc[mt], false, false);
        }
    };

    // One GEMM phase over nkb (even) k-blocks; tile for kb=0 already staged
    // into buf0. nextA != nullptr => peeled epilogue stages the next phase's
    // first tile so the pipeline never drains at the phase boundary.
    auto run_phase = [&](const bf16_t* Athr, const bf16_t* Bthr, int nkb,
                         const bf16_t* nextA) {
        // steady state: unconditional stage of kb+1, branch-free
#pragma unroll 2
        for (int kb = 0; kb < nkb - 1; ++kb) {
            async_ld_b128(ldst + ((kb & 1) ^ 1) * 4096, Athr + (kb + 1) * 32);
            __builtin_prefetch(Bthr + (kb + 1) * 32, 0, 3);
            v16bf bf = frag_from(Bthr + kb * 32);   // B frags straight from L2
            wait_async_le1();                        // tile kb landed
            __syncthreads();                         // visible block-wide
            compute_blk(kb & 1, bf);
            __syncthreads();                         // done reading buf[kb&1]
        }
        // peeled last block: kb = nkb-1, cur = 1 (nkb even), stages into buf0
        if (nextA) async_ld_b128(ldst, nextA);
        v16bf bf = frag_from(Bthr + (nkb - 1) * 32);
        if (nextA) wait_async_le1(); else wait_async_0();
        __syncthreads();
        compute_blk(1, bf);
        __syncthreads();
    };

    async_ld_b128(ldst, Ax);                 // prologue: first x tile
    run_phase(Ax, Bw0, Kin >> 5, Ah);        // x @ Wih^T  (prefetches h tile)
    run_phase(Ah, Bw1, HID >> 5, nullptr);   // h @ Whh^T

    // Bias + activation -> LDS gate buffer [gate][m][ncol]
    {
        const int ncol = nsub + m16;                // 0..31
        const int j    = gate * HID + nb * 32 + ncol;
        const float bsum = b_ih[j] + b_hh[j];
#pragma unroll
        for (int mt = 0; mt < 4; ++mt) {
#pragma unroll
            for (int r = 0; r < 8; ++r) {
                const int mrow = mt * 16 + (gsel ? (r + 8) : r);
                float v = acc[mt][r] + bsum;
                v = (gate == 2) ? tanhf(v) : sigmoidf_(v);
                smem_f[gate * (BATCH * 32) + mrow * 32 + ncol] = v;
            }
        }
    }
    __syncthreads();

    // Elementwise cell: 64x32 elements, 8 per thread.
#pragma unroll
    for (int ii = 0; ii < 8; ++ii) {
        const int e = tid + ii * 256;        // 0..2047
        const int m = e >> 5;
        const int n = e & 31;
        const float ig = smem_f[0 * 2048 + e];
        const float fg = smem_f[1 * 2048 + e];
        const float gg = smem_f[2 * 2048 + e];
        const float og = smem_f[3 * 2048 + e];
        const int j  = nb * 32 + n;
        const int ci = m * HID + j;
        const float cn = fg * cst[ci] + ig * gg;
        cst[ci]  = cn;
        hout[ci] = (bf16_t)(og * tanhf(cn));
    }
}

// ---------------------------------------------------------------------------
// y = relu(h @ Wfc^T + b_fc); store fp32 slice of (B, SEQ, OUT) output and
// bf16 copy as next-step decoder input. Grid: 4 blocks x 128 columns.
// ---------------------------------------------------------------------------
__global__ __launch_bounds__(256)
void fc_relu_kernel(const bf16_t* __restrict__ hmat,  // (64, 1024)
                    const bf16_t* __restrict__ Wfc,   // (512, 1024)
                    const float*  __restrict__ bfc,   // (512)
                    float*        __restrict__ out,   // (64, 256, 512)
                    int t,
                    bf16_t*       __restrict__ xnext) // (64, 512)
{
    __shared__ __align__(16) bf16_t sA[2 * BATCH * 32];   // 8 KB double buffer

    const int tid  = threadIdx.x;
    const int wave = tid >> 5;
    const int lane = tid & 31;
    const int m16  = lane & 15;
    const int gsel = lane >> 4;
    const int colbase = blockIdx.x * 128 + wave * 16;

    const uint32_t sA_lds = (uint32_t)(size_t)(void*)sA;

    v8f acc[4];
#pragma unroll
    for (int mt = 0; mt < 4; ++mt)
#pragma unroll
        for (int r = 0; r < 8; ++r) acc[mt][r] = 0.0f;

    const int arow = tid >> 2;
    const int acol = tid & 3;
    const int nkb  = HID >> 5;

    const bf16_t* Athr = hmat + arow * HID + acol * 8;
    const bf16_t* Bthr = Wfc + (size_t)(colbase + m16) * HID + gsel * 8;
    const uint32_t ldst = sA_lds + tid * 16;

    auto compute_blk = [&](int cur, v16bf bf) {
        const bf16_t* abase = sA + cur * 2048 + m16 * 32 + gsel * 8;
#pragma unroll
        for (int mt = 0; mt < 4; ++mt) {
            v16bf af = frag_from(abase + mt * 16 * 32);
            acc[mt] = __builtin_amdgcn_wmma_f32_16x16x32_bf16(
                false, af, false, bf, (short)0, acc[mt], false, false);
        }
    };

    async_ld_b128(ldst, Athr);
#pragma unroll 2
    for (int kb = 0; kb < nkb - 1; ++kb) {
        async_ld_b128(ldst + ((kb & 1) ^ 1) * 4096, Athr + (kb + 1) * 32);
        __builtin_prefetch(Bthr + (kb + 1) * 32, 0, 3);
        v16bf bf = frag_from(Bthr + kb * 32);
        wait_async_le1();
        __syncthreads();
        compute_blk(kb & 1, bf);
        __syncthreads();
    }
    {   // peeled last block (cur = 1, nkb even)
        v16bf bf = frag_from(Bthr + (nkb - 1) * 32);
        wait_async_0();
        __syncthreads();
        compute_blk(1, bf);
        __syncthreads();
    }

    const int col = colbase + m16;
    const float b = bfc[col];
#pragma unroll
    for (int mt = 0; mt < 4; ++mt) {
#pragma unroll
        for (int r = 0; r < 8; ++r) {
            const int row = mt * 16 + (gsel ? (r + 8) : r);
            float v = acc[mt][r] + b;
            v = fmaxf(v, 0.0f);
            out[((size_t)row * SEQL + t) * OUTD + col] = v;
            xnext[row * OUTD + col] = (bf16_t)v;
        }
    }
}

// --------------------------- prep kernels ----------------------------------
__global__ void cvt_bf16_kernel(const float* __restrict__ src,
                                bf16_t* __restrict__ dst, int n) {
    int i = blockIdx.x * blockDim.x + threadIdx.x;
    if (i < n) dst[i] = (bf16_t)src[i];
}

__global__ void fill_bf16_kernel(bf16_t* __restrict__ dst, float val, int n) {
    int i = blockIdx.x * blockDim.x + threadIdx.x;
    if (i < n) dst[i] = (bf16_t)val;
}

__global__ void copy_f32_kernel(const float* __restrict__ src,
                                float* __restrict__ dst, int n) {
    int i = blockIdx.x * blockDim.x + threadIdx.x;
    if (i < n) dst[i] = src[i];
}

// ---------------------------------------------------------------------------
extern "C" void kernel_launch(void* const* d_in, const int* in_sizes, int n_in,
                              void* d_out, int out_size, void* d_ws, size_t ws_size,
                              hipStream_t stream) {
    (void)in_sizes; (void)n_in; (void)out_size; (void)ws_size;

    const float* h0    = (const float*)d_in[1];   // (2, 64, 1024)
    const float* c0    = (const float*)d_in[2];   // (2, 64, 1024)
    const float* W_ih0 = (const float*)d_in[3];   // (4096, 512)
    const float* W_hh0 = (const float*)d_in[4];   // (4096, 1024)
    const float* b_ih0 = (const float*)d_in[5];
    const float* b_hh0 = (const float*)d_in[6];
    const float* W_ih1 = (const float*)d_in[7];   // (4096, 1024)
    const float* W_hh1 = (const float*)d_in[8];   // (4096, 1024)
    const float* b_ih1 = (const float*)d_in[9];
    const float* b_hh1 = (const float*)d_in[10];
    const float* W_fc  = (const float*)d_in[11];  // (512, 1024)
    const float* b_fc  = (const float*)d_in[12];
    float* out = (float*)d_out;

    // ---- workspace carve-up (bf16 weights stay L2-resident: ~29 MB) ----
    char* ws = (char*)d_ws;
    size_t off = 0;
    auto carve = [&](size_t bytes) -> void* {
        void* p = ws + off;
        off = (off + bytes + 255) & ~(size_t)255;
        return p;
    };
    bf16_t* Wih0b = (bf16_t*)carve((size_t)NGATE * 512 * 2);
    bf16_t* Whh0b = (bf16_t*)carve((size_t)NGATE * HID * 2);
    bf16_t* Wih1b = (bf16_t*)carve((size_t)NGATE * HID * 2);
    bf16_t* Whh1b = (bf16_t*)carve((size_t)NGATE * HID * 2);
    bf16_t* Wfcb  = (bf16_t*)carve((size_t)OUTD * HID * 2);
    bf16_t* xbf   = (bf16_t*)carve((size_t)BATCH * OUTD * 2);
    bf16_t* h0buf[2] = { (bf16_t*)carve((size_t)BATCH * HID * 2),
                         (bf16_t*)carve((size_t)BATCH * HID * 2) };
    bf16_t* h1buf[2] = { (bf16_t*)carve((size_t)BATCH * HID * 2),
                         (bf16_t*)carve((size_t)BATCH * HID * 2) };
    float* c0s = (float*)carve((size_t)BATCH * HID * 4);
    float* c1s = (float*)carve((size_t)BATCH * HID * 4);

    auto cvt = [&](const float* s, bf16_t* d, int n) {
        cvt_bf16_kernel<<<(n + 255) / 256, 256, 0, stream>>>(s, d, n);
    };

    // ---- per-call (deterministic) state init ----
    cvt(W_ih0, Wih0b, NGATE * 512);
    cvt(W_hh0, Whh0b, NGATE * HID);
    cvt(W_ih1, Wih1b, NGATE * HID);
    cvt(W_hh1, Whh1b, NGATE * HID);
    cvt(W_fc,  Wfcb,  OUTD * HID);
    cvt(h0,                 h0buf[0], BATCH * HID);    // layer 0 h init
    cvt(h0 + BATCH * HID,   h1buf[0], BATCH * HID);    // layer 1 h init
    copy_f32_kernel<<<(BATCH * HID + 255) / 256, 256, 0, stream>>>(
        c0, c0s, BATCH * HID);
    copy_f32_kernel<<<(BATCH * HID + 255) / 256, 256, 0, stream>>>(
        c0 + BATCH * HID, c1s, BATCH * HID);
    fill_bf16_kernel<<<(BATCH * OUTD + 255) / 256, 256, 0, stream>>>(
        xbf, -2.0f, BATCH * OUTD);   // SOS token

    // ---- 256 sequential decoder steps ----
    int cur = 0;
    for (int t = 0; t < SEQL; ++t) {
        const int nxt = cur ^ 1;
        gate_cell_kernel<<<32, 256, 0, stream>>>(
            xbf, OUTD, h0buf[cur], Wih0b, Whh0b, b_ih0, b_hh0, c0s, h0buf[nxt]);
        gate_cell_kernel<<<32, 256, 0, stream>>>(
            h0buf[nxt], HID, h1buf[cur], Wih1b, Whh1b, b_ih1, b_hh1, c1s, h1buf[nxt]);
        fc_relu_kernel<<<4, 256, 0, stream>>>(
            h1buf[nxt], Wfcb, b_fc, out, t, xbf);
        cur = nxt;
    }
}
